// HeteroAttention_17394617548986
// MI455X (gfx1250) — matrix operand order repaired and checked by reading, not compiled, not run on hardware
//
#include <hip/hip_runtime.h>
#include <hip/hip_bf16.h>
#include <math.h>

// ---------------- problem constants ----------------
#define DIMD     256
#define DH       32
#define NHEADS   8
#define NTYPES   2
#define WS       7
#define WW       49          // WS*WS window positions
#define AGENTS   6
#define NB       4
#define XY       64          // 8*8 windows
#define TOK      3136        // XY*WW tokens per (b,l)
#define KEYS     294         // AGENTS*WW
#define PADK     320         // KEYS padded to 20 tiles of 16 (10 K-steps of 32)
#define MTILES   196         // TOK/16
#define SCALE_Q  0.17677669529663687f   // 32^-0.5

typedef __attribute__((ext_vector_type(16))) __bf16 v16bf;
typedef __attribute__((ext_vector_type(8)))  float  v8f;

// A-matrix (16x32, bf16) element K-offset for half i of this lane.
// lanes 0-15: K in {0..7} U {16..23}; lanes 16-31: K in {8..15} U {24..31}; M = lane&15
__device__ __forceinline__ int a_koff(int lane, int i) {
    return ((lane < 16) ? 0 : 8) + ((i < 8) ? i : (i + 8));
}
// B-matrix (32x16, bf16): lanes 0-15: K in {0..7} U {16..23}; lanes 16-31: K in {8..15} U {24..31}; N = lane&15
__device__ __forceinline__ int b_koff(int lane, int i) {
    return ((lane < 16) ? 0 : 8) + ((i < 8) ? i : (i + 8));
}

#define WMMA_BF16(A, Bf, C) \
    __builtin_amdgcn_wmma_f32_16x16x32_bf16(false, (A), false, (Bf), (short)0, (C), false, false)

// ============================================================
// Kernel 1a: K/V projection for all (b,l).  y = x * W^T + b
// Register-blocked 16x64 tile per wave; branch-free hot loop.
// grid = (MTILES*4, B*AGENTS), block = 32
// ============================================================
__global__ __launch_bounds__(32) void kv_proj_kernel(
    const float* __restrict__ x, const int* __restrict__ mode,
    const float* __restrict__ Wk, const float* __restrict__ bk,
    const float* __restrict__ Wv, const float* __restrict__ bv,
    float* __restrict__ kwbuf, float* __restrict__ vmbuf)
{
    const int tile = blockIdx.x;
    const int mt = tile >> 2, nt64 = tile & 3;
    const int bl = blockIdx.y;
    const int lane = threadIdx.x;
    const int md = mode[bl];

    const float* xa  = x  + (size_t)bl * TOK * DIMD;
    const float* wkB = Wk + (size_t)md * DIMD * DIMD;
    const float* wvB = Wv + (size_t)md * DIMD * DIMD;

    const int mrow = mt * 16 + (lane & 15);
    const int nl   = lane & 15;

    v8f acck[4] = {};
    v8f accv[4] = {};

    for (int kk = 0; kk < DIMD / 32; ++kk) {
        const int kbase = kk * 32;
        const float* arow = xa + (size_t)mrow * DIMD + kbase;
        __builtin_prefetch(arow + 32, 0, 0);   // next K-chunk of streamed A row
        v16bf a;
#pragma unroll
        for (int i = 0; i < 16; ++i) a[i] = (__bf16)arow[a_koff(lane, i)];

#pragma unroll
        for (int j = 0; j < 4; ++j) {
            const size_t wrow = (size_t)(nt64 * 64 + j * 16 + nl) * DIMD + kbase;
            v16bf fk, fv;
#pragma unroll
            for (int i = 0; i < 16; ++i) {
                const int k = b_koff(lane, i);
                fk[i] = (__bf16)wkB[wrow + k];
                fv[i] = (__bf16)wvB[wrow + k];
            }
            acck[j] = WMMA_BF16(a, fk, acck[j]);
            accv[j] = WMMA_BF16(a, fv, accv[j]);
        }
    }

    // C/D layout: N = lane&15, M = r + (lane<16 ? 0 : 8)
    const int m0 = mt * 16 + ((lane < 16) ? 0 : 8);
#pragma unroll
    for (int j = 0; j < 4; ++j) {
        const int ncol = nt64 * 64 + j * 16 + nl;
        const float bkv = bk[md * DIMD + ncol];
        const float bvv = bv[md * DIMD + ncol];
#pragma unroll
        for (int r = 0; r < 8; ++r) {
            const int m = m0 + r;
            const size_t o = ((size_t)bl * TOK + m) * DIMD + ncol;
            kwbuf[o] = acck[j][r] + bkv;
            vmbuf[o] = accv[j][r] + bvv;
        }
    }
}

// ============================================================
// Kernel 1b: Q projection, agent 0 only.  q = (x * Wq^T + bq) * SCALE
// grid = (MTILES*4, NB), block = 32
// ============================================================
__global__ __launch_bounds__(32) void q_proj_kernel(
    const float* __restrict__ x, const int* __restrict__ mode,
    const float* __restrict__ Wq, const float* __restrict__ bq,
    float* __restrict__ qbuf)
{
    const int tile = blockIdx.x;
    const int mt = tile >> 2, nt64 = tile & 3;
    const int b = blockIdx.y;
    const int lane = threadIdx.x;
    const int md = mode[b * AGENTS];       // agent 0's mode

    const float* xa  = x  + (size_t)(b * AGENTS) * TOK * DIMD;   // l = 0 slice
    const float* wqB = Wq + (size_t)md * DIMD * DIMD;

    const int mrow = mt * 16 + (lane & 15);
    const int nl   = lane & 15;

    v8f accq[4] = {};
    for (int kk = 0; kk < DIMD / 32; ++kk) {
        const int kbase = kk * 32;
        const float* arow = xa + (size_t)mrow * DIMD + kbase;
        __builtin_prefetch(arow + 32, 0, 0);
        v16bf a;
#pragma unroll
        for (int i = 0; i < 16; ++i) a[i] = (__bf16)arow[a_koff(lane, i)];
#pragma unroll
        for (int j = 0; j < 4; ++j) {
            const size_t wrow = (size_t)(nt64 * 64 + j * 16 + nl) * DIMD + kbase;
            v16bf fq;
#pragma unroll
            for (int i = 0; i < 16; ++i) fq[i] = (__bf16)wqB[wrow + b_koff(lane, i)];
            accq[j] = WMMA_BF16(a, fq, accq[j]);
        }
    }
    const int m0 = mt * 16 + ((lane < 16) ? 0 : 8);
#pragma unroll
    for (int j = 0; j < 4; ++j) {
        const int ncol = nt64 * 64 + j * 16 + nl;
        const float bqv = bq[md * DIMD + ncol];
#pragma unroll
        for (int r = 0; r < 8; ++r) {
            const int m = m0 + r;
            qbuf[((size_t)b * TOK + m) * DIMD + ncol] = (accq[j][r] + bqv) * SCALE_Q;
        }
    }
}

// ============================================================
// Kernel 2: per-head 32x32 relation transforms, in place.
// kw = k * w_att^T ; vmsg = v * w_msg    (per (b,l,head))
// grid = (MTILES, NHEADS, B*AGENTS), block = 32
// ============================================================
__global__ __launch_bounds__(32) void head_transform_kernel(
    const int* __restrict__ mode,
    const float* __restrict__ rel_att, const float* __restrict__ rel_msg,
    float* __restrict__ kwbuf, float* __restrict__ vmbuf)
{
    const int mt = blockIdx.x;
    const int h  = blockIdx.y;
    const int bl = blockIdx.z;
    const int b  = bl / AGENTS;
    const int lane = threadIdx.x;

    const int rel = mode[b * AGENTS] * NTYPES + mode[bl];
    const float* wa = rel_att + (((size_t)rel * NHEADS) + h) * DH * DH;
    const float* wm = rel_msg + (((size_t)rel * NHEADS) + h) * DH * DH;

    float* kp = kwbuf + ((size_t)bl * TOK + mt * 16) * DIMD + h * DH;
    float* vp = vmbuf + ((size_t)bl * TOK + mt * 16) * DIMD + h * DH;

    const int mloc = lane & 15;
    v16bf ak, av;
#pragma unroll
    for (int i = 0; i < 16; ++i) {
        const int k = a_koff(lane, i);
        ak[i] = (__bf16)kp[(size_t)mloc * DIMD + k];
        av[i] = (__bf16)vp[(size_t)mloc * DIMD + k];
    }
    const int nl = lane & 15;
    v16bf ba0, ba1, bm0, bm1;
#pragma unroll
    for (int i = 0; i < 16; ++i) {
        const int k = b_koff(lane, i);
        ba0[i] = (__bf16)wa[(size_t)nl * DH + k];          // B[k][n] = w_att[n][k]
        ba1[i] = (__bf16)wa[(size_t)(nl + 16) * DH + k];
        bm0[i] = (__bf16)wm[(size_t)k * DH + nl];          // B[k][n] = w_msg[k][n]
        bm1[i] = (__bf16)wm[(size_t)k * DH + nl + 16];
    }
    v8f k0 = {}, k1 = {}, v0 = {}, v1 = {};
    k0 = WMMA_BF16(ak, ba0, k0);
    k1 = WMMA_BF16(ak, ba1, k1);
    v0 = WMMA_BF16(av, bm0, v0);
    v1 = WMMA_BF16(av, bm1, v1);

    const int m0 = (lane < 16) ? 0 : 8;
#pragma unroll
    for (int r = 0; r < 8; ++r) {
        const size_t row = (size_t)(m0 + r) * DIMD;
        kp[row + nl]      = k0[r];
        kp[row + nl + 16] = k1[r];
        vp[row + nl]      = v0[r];
        vp[row + nl + 16] = v1[r];
    }
}

// ============================================================
// Kernel 3: windowed attention per (b, window, head).
// S = Q*KW^T (+bias), softmax over 294 keys, O = P*Vmsg.
// grid = (XY, NHEADS, NB), block = 128 (4 waves); S in LDS (49x320 f32 ~ 61 KB)
// ============================================================
__global__ __launch_bounds__(128) void attention_kernel(
    const float* __restrict__ qbuf, const float* __restrict__ kwbuf,
    const float* __restrict__ vmbuf, const float* __restrict__ pos_table,
    float* __restrict__ obuf)
{
    __shared__ float S[WW * PADK];

    const int xy = blockIdx.x, h = blockIdx.y, b = blockIdx.z;
    const int lane = threadIdx.x & 31;
    const int wv   = threadIdx.x >> 5;
    const int nl   = lane & 15;

    const float* qp = qbuf + ((size_t)b * TOK + xy * WW) * DIMD + h * DH;

    // ---- phase 1: S tiles = Q(16x32) x KW^T(32x16) ----
    // Each wave owns 5 fixed N-tiles (nt = wv + 4j); B fragments preloaded once.
    v16bf bfr[5];
#pragma unroll
    for (int j = 0; j < 5; ++j) {
        const int nt  = wv + 4 * j;
        const int key = nt * 16 + nl;
        const bool kok = (key < KEYS);
        const int z = key / WW, ef = key % WW;
        const float* kp = kwbuf + ((size_t)(b * AGENTS + z) * TOK + xy * WW + ef) * DIMD + h * DH;
#pragma unroll
        for (int i = 0; i < 16; ++i)
            bfr[j][i] = kok ? (__bf16)kp[b_koff(lane, i)] : (__bf16)0.0f;
    }
    for (int mt = 0; mt < 4; ++mt) {
        const int mrow = mt * 16 + nl;
        const bool mok = (mrow < WW);
        v16bf a;
#pragma unroll
        for (int i = 0; i < 16; ++i)
            a[i] = mok ? (__bf16)qp[(size_t)mrow * DIMD + a_koff(lane, i)] : (__bf16)0.0f;
        const int m0 = mt * 16 + ((lane < 16) ? 0 : 8);
#pragma unroll
        for (int j = 0; j < 5; ++j) {
            v8f acc = {};
            acc = WMMA_BF16(a, bfr[j], acc);
            const int nt = wv + 4 * j;
#pragma unroll
            for (int r = 0; r < 8; ++r) {
                const int m = m0 + r;
                if (m < WW) S[m * PADK + nt * 16 + nl] = acc[r];
            }
        }
    }
    __syncthreads();

    // ---- phase 2: +relative-position bias, softmax over 294 keys per row ----
    for (int row = threadIdx.x; row < WW; row += 128) {
        const int qi = row / WS, qj = row % WS;
        float mx = -1e30f;
        for (int kpos = 0; kpos < KEYS; ++kpos) {
            const int ef = kpos % WW;
            const int ki = ef / WS, kj = ef % WS;
            const int idx = (qi - ki + WS - 1) * (2 * WS - 1) + (qj - kj + WS - 1);
            const float val = S[row * PADK + kpos] + pos_table[idx * NHEADS + h];
            S[row * PADK + kpos] = val;
            mx = fmaxf(mx, val);
        }
        float sum = 0.0f;
        for (int kpos = 0; kpos < KEYS; ++kpos) {
            const float e = __expf(S[row * PADK + kpos] - mx);
            S[row * PADK + kpos] = e;
            sum += e;
        }
        const float inv = 1.0f / sum;
        for (int kpos = 0; kpos < KEYS; ++kpos) S[row * PADK + kpos] *= inv;
        for (int kpos = KEYS; kpos < PADK; ++kpos) S[row * PADK + kpos] = 0.0f;
    }
    __syncthreads();

    // ---- phase 3: O = P(49x320) x Vmsg(320x32) ----
    // Each wave owns one 16-col output slice (nc0) and two M-tiles; B shared.
    {
        const int nc0 = (wv & 1) * 16;
        const int mtA = wv >> 1;          // 0..1
        const int mtB = mtA + 2;          // 2..3
        const int mrowA = mtA * 16 + nl;
        const int mrowB = mtB * 16 + nl;
        const bool mokA = (mrowA < WW);
        const bool mokB = (mrowB < WW);
        v8f accA = {}, accB = {};
        for (int kk = 0; kk < PADK / 32; ++kk) {
            v16bf bb;
#pragma unroll
            for (int i = 0; i < 16; ++i) {
                const int key = kk * 32 + b_koff(lane, i);
                const bool kok = (key < KEYS);
                const int z = key / WW, ef = key % WW;
                const float* vp = vmbuf +
                    ((size_t)(b * AGENTS + z) * TOK + xy * WW + ef) * DIMD + h * DH + nc0 + nl;
                bb[i] = kok ? (__bf16)vp[0] : (__bf16)0.0f;
            }
            v16bf aA, aB;
#pragma unroll
            for (int i = 0; i < 16; ++i) {
                const int key = kk * 32 + a_koff(lane, i);
                aA[i] = mokA ? (__bf16)S[mrowA * PADK + key] : (__bf16)0.0f;
                aB[i] = mokB ? (__bf16)S[mrowB * PADK + key] : (__bf16)0.0f;
            }
            accA = WMMA_BF16(aA, bb, accA);
            accB = WMMA_BF16(aB, bb, accB);
        }
        const int mb = (lane < 16) ? 0 : 8;
#pragma unroll
        for (int r = 0; r < 8; ++r) {
            const int mA = mtA * 16 + mb + r;
            const int mB = mtB * 16 + mb + r;
            if (mA < WW)
                obuf[((size_t)b * TOK + xy * WW + mA) * DIMD + h * DH + nc0 + nl] = accA[r];
            if (mB < WW)
                obuf[((size_t)b * TOK + xy * WW + mB) * DIMD + h * DH + nc0 + nl] = accB[r];
        }
    }
}

// ============================================================
// Kernel 4: output projection.  out = O * Wa[mode0]^T + ba[mode0]
// Register-blocked 16x64 tile per wave.
// grid = (MTILES*4, NB), block = 32
// ============================================================
__global__ __launch_bounds__(32) void out_proj_kernel(
    const float* __restrict__ obuf, const int* __restrict__ mode,
    const float* __restrict__ Wa, const float* __restrict__ ba,
    float* __restrict__ out)
{
    const int tile = blockIdx.x;
    const int mt = tile >> 2, nt64 = tile & 3;
    const int b = blockIdx.y;
    const int lane = threadIdx.x;
    const int md = mode[b * AGENTS];   // mode of agent 0

    const float* xa = obuf + (size_t)b * TOK * DIMD;
    const float* wB = Wa + (size_t)md * DIMD * DIMD;

    const int mrow = mt * 16 + (lane & 15);
    const int nl   = lane & 15;

    v8f acc[4] = {};
    for (int kk = 0; kk < DIMD / 32; ++kk) {
        const int kbase = kk * 32;
        const float* arow = xa + (size_t)mrow * DIMD + kbase;
        __builtin_prefetch(arow + 32, 0, 0);
        v16bf a;
#pragma unroll
        for (int i = 0; i < 16; ++i) a[i] = (__bf16)arow[a_koff(lane, i)];
#pragma unroll
        for (int j = 0; j < 4; ++j) {
            const size_t wrow = (size_t)(nt64 * 64 + j * 16 + nl) * DIMD + kbase;
            v16bf bb;
#pragma unroll
            for (int i = 0; i < 16; ++i) bb[i] = (__bf16)wB[wrow + b_koff(lane, i)];
            acc[j] = WMMA_BF16(a, bb, acc[j]);
        }
    }
    const int m0 = mt * 16 + ((lane < 16) ? 0 : 8);
#pragma unroll
    for (int j = 0; j < 4; ++j) {
        const int ncol = nt64 * 64 + j * 16 + nl;
        const float bav = ba[md * DIMD + ncol];
#pragma unroll
        for (int r = 0; r < 8; ++r) {
            const int m = m0 + r;
            out[((size_t)b * TOK + m) * DIMD + ncol] = acc[j][r] + bav;
        }
    }
}

// ============================================================
extern "C" void kernel_launch(void* const* d_in, const int* in_sizes, int n_in,
                              void* d_out, int out_size, void* d_ws, size_t ws_size,
                              hipStream_t stream) {
    const float* x    = (const float*)d_in[0];
    const int*   mode = (const int*)  d_in[1];
    const float* Wq   = (const float*)d_in[2];
    const float* bq   = (const float*)d_in[3];
    const float* Wk   = (const float*)d_in[4];
    const float* bk   = (const float*)d_in[5];
    const float* Wv   = (const float*)d_in[6];
    const float* bv   = (const float*)d_in[7];
    const float* Wa   = (const float*)d_in[8];
    const float* ba   = (const float*)d_in[9];
    const float* ratt = (const float*)d_in[10];
    const float* rmsg = (const float*)d_in[11];
    const float* ptab = (const float*)d_in[12];
    float* out = (float*)d_out;

    float* ws    = (float*)d_ws;
    float* qbuf  = ws;                                  // B*TOK*DIMD
    float* kwbuf = qbuf  + (size_t)NB * TOK * DIMD;     // B*L*TOK*DIMD (k -> kw in place)
    float* vmbuf = kwbuf + (size_t)NB * AGENTS * TOK * DIMD;
    float* obuf  = vmbuf + (size_t)NB * AGENTS * TOK * DIMD;

    kv_proj_kernel<<<dim3(MTILES * 4, NB * AGENTS), 32, 0, stream>>>(
        x, mode, Wk, bk, Wv, bv, kwbuf, vmbuf);

    q_proj_kernel<<<dim3(MTILES * 4, NB), 32, 0, stream>>>(
        x, mode, Wq, bq, qbuf);

    head_transform_kernel<<<dim3(MTILES, NHEADS, NB * AGENTS), 32, 0, stream>>>(
        mode, ratt, rmsg, kwbuf, vmbuf);

    attention_kernel<<<dim3(XY, NHEADS, NB), 128, 0, stream>>>(
        qbuf, kwbuf, vmbuf, ptab, obuf);

    out_proj_kernel<<<dim3(MTILES * 4, NB), 32, 0, stream>>>(
        obuf, mode, Wa, ba, out);
}
